// CVIT_89275190214827
// MI455X (gfx1250) — matrix operand order repaired and checked by reading, not compiled, not run on hardware
//
#include <hip/hip_runtime.h>

// ---------------------------------------------------------------------------
// MI455X (gfx1250): f16 WMMA GEMMs (f32 accum) for all K=256 matmuls.
// Each block computes a 64-row x full-N strip so the (huge) A operand is
// streamed from HBM exactly once; weights stay hot in L2. Wave32 VALU kernels
// handle deformable sampling / softmax / tiny attention.
// ---------------------------------------------------------------------------

typedef __attribute__((ext_vector_type(16))) _Float16 v16h;
typedef __attribute__((ext_vector_type(8)))  _Float16 v8h;
typedef __attribute__((ext_vector_type(4)))  _Float16 v4h;
typedef __attribute__((ext_vector_type(8)))  float    v8f;

namespace {
constexpr int kB    = 32;
constexpr int kD    = 256;
constexpr int kNH   = 8;
constexpr int kL    = 3;
constexpr int kP    = 4;
constexpr int kDH   = 32;      // kD / kNH
constexpr int kLQ   = 1024;    // 32*32
constexpr int kLIN  = 5376;    // 64*64 + 32*32 + 16*16
constexpr int kNCLS = 200;
constexpr int kNC   = 10;
constexpr int kMQ   = kB * kLQ;    // 32768 query rows
constexpr int kMV   = kB * kLIN;   // 172032 value rows
constexpr float kInvSqrt128 = 0.08838834764831845f;
}

// ---------------------------------------------------------------------------
// Strip GEMM: C[M,N] = A[M,K] * Bw[K,N] (+bias[n]) (+resid[m,n])
// Template NT = number of 16-wide column tiles (N <= NT*16 < N+16).
// Block = 128 threads (4 waves); block computes rows [m0, m0+64) x all N.
// Per K-step each wave loads ONE A fragment and runs NT WMMAs against it.
// REQUIREMENTS: M % 64 == 0, K % 32 == 0 (true for all launches here).
// ---------------------------------------------------------------------------
template <int NT>
__global__ __launch_bounds__(128)
void gemm_strip(const float* __restrict__ A, const float* __restrict__ Bw,
                const float* __restrict__ bias, const float* __restrict__ resid,
                float* __restrict__ C, int M, int N, int K, int ldc)
{
    constexpr int BKP = 40;                    // padded K stride (halves)
    constexpr int BN  = NT * 16;
    __shared__ _Float16 Ah[64][BKP];           // [m][k]
    __shared__ _Float16 Bh[BN][BKP];           // [n][k]  (transposed weight strip)

    const int t    = threadIdx.x;
    const int lane = t & 31;
    const int wave = t >> 5;
    const int half = lane >> 4;                // 0: lanes 0-15, 1: lanes 16-31
    const int l16  = lane & 15;
    const int m0   = blockIdx.x * 64;
    const bool fullN = (BN <= N);              // compile-time-ish, uniform

    v8f acc[NT];
#pragma unroll
    for (int j = 0; j < NT; ++j)
#pragma unroll
        for (int r = 0; r < 8; ++r) acc[j][r] = 0.0f;

    // A staging map (float4): idx = i*128+t -> m = idx>>3 (0..63), c = (idx&7)*4
    float4 aReg[4];
#pragma unroll
    for (int i = 0; i < 4; ++i) {
        const int idx = i * 128 + t;
        const int m = idx >> 3, c = (idx & 7) * 4;
        aReg[i] = *(const float4*)(A + (size_t)(m0 + m) * K + c);
    }

    for (int k0 = 0; k0 < K; k0 += 32) {
        // ---- store staged A registers to LDS (one b64 per chunk) ----
#pragma unroll
        for (int i = 0; i < 4; ++i) {
            const int idx = i * 128 + t;
            const int m = idx >> 3, c = (idx & 7) * 4;
            v4h hv;
            hv[0] = (_Float16)aReg[i].x; hv[1] = (_Float16)aReg[i].y;
            hv[2] = (_Float16)aReg[i].z; hv[3] = (_Float16)aReg[i].w;
            *(v4h*)(&Ah[m][c]) = hv;
        }
        // ---- stage B strip (32 x BN) transposed to [n][k] ----
        // chunk map: c = i*128+t -> kk = c/(NT*4), nn = (c%(NT*4))*4
        if (fullN) {
#pragma unroll
            for (int i = 0; i < NT; ++i) {
                const int cidx = i * 128 + t;
                const int kk = cidx / (NT * 4), nn = (cidx % (NT * 4)) * 4;
                const float4 v = *(const float4*)(Bw + (size_t)(k0 + kk) * N + nn);
                Bh[nn + 0][kk] = (_Float16)v.x;
                Bh[nn + 1][kk] = (_Float16)v.y;
                Bh[nn + 2][kk] = (_Float16)v.z;
                Bh[nn + 3][kk] = (_Float16)v.w;
            }
        } else {
            // padded columns (e.g. N=200, NT=13): branchless clamp + select
#pragma unroll
            for (int i = 0; i < NT; ++i) {
                const int cidx = i * 128 + t;
                const int kk = cidx / (NT * 4), nn = (cidx % (NT * 4)) * 4;
                const float* p = Bw + (size_t)(k0 + kk) * N;
                float e0 = p[min(nn + 0, N - 1)];
                float e1 = p[min(nn + 1, N - 1)];
                float e2 = p[min(nn + 2, N - 1)];
                float e3 = p[min(nn + 3, N - 1)];
                Bh[nn + 0][kk] = (_Float16)((nn + 0 < N) ? e0 : 0.0f);
                Bh[nn + 1][kk] = (_Float16)((nn + 1 < N) ? e1 : 0.0f);
                Bh[nn + 2][kk] = (_Float16)((nn + 2 < N) ? e2 : 0.0f);
                Bh[nn + 3][kk] = (_Float16)((nn + 3 < N) ? e3 : 0.0f);
            }
        }
        __syncthreads();

        // ---- prefetch next A strip into registers (overlaps WMMA) ----
        if (k0 + 32 < K) {
#pragma unroll
            for (int i = 0; i < 4; ++i) {
                const int idx = i * 128 + t;
                const int m = idx >> 3, c = (idx & 7) * 4;
                aReg[i] = *(const float4*)(A + (size_t)(m0 + m) * K + (k0 + 32 + c));
            }
        }

        // ---- A fragment per ISA 7.12.2 (16-bit A 16x32):
        //      lanes 0-15: halves[0..7]=K0..7,  halves[8..15]=K16..23
        //      lanes 16-31: halves[0..7]=K8..15, halves[8..15]=K24..31
        const int rowA = (wave << 4) + l16;
        v8h alo = *(const v8h*)(&Ah[rowA][half ? 8 : 0]);
        v8h ahi = *(const v8h*)(&Ah[rowA][half ? 24 : 16]);
        v16h afrag;
#pragma unroll
        for (int r = 0; r < 8; ++r) { afrag[r] = alo[r]; afrag[8 + r] = ahi[r]; }

#pragma unroll
        for (int j = 0; j < NT; ++j) {
            // B fragment: lanes 0-15 col=lane, K0..7,K16..23; lanes 16-31 K8..15,K24..31
            const int colB = j * 16 + l16;
            v8h blo = *(const v8h*)(&Bh[colB][half ? 8 : 0]);
            v8h bhi = *(const v8h*)(&Bh[colB][half ? 24 : 16]);
            v16h bfrag;
#pragma unroll
            for (int r = 0; r < 8; ++r) { bfrag[r] = blo[r]; bfrag[8 + r] = bhi[r]; }
            acc[j] = __builtin_amdgcn_wmma_f32_16x16x32_f16(
                false, afrag, false, bfrag, (short)0, acc[j], false, false);
        }
        __syncthreads();
    }

    // ---- store: C/D layout: VGPR r -> (M=r, N=lane) / (M=8+r, N=lane-16) ----
#pragma unroll
    for (int j = 0; j < NT; ++j) {
        const int n = j * 16 + l16;
        if (fullN || n < N) {
            const float bv = bias ? bias[n] : 0.0f;
#pragma unroll
            for (int r = 0; r < 8; ++r) {
                const int m = m0 + (wave << 4) + (half ? 8 : 0) + r;
                float v = acc[j][r] + bv;
                if (resid) v += resid[(size_t)m * ldc + n];
                C[(size_t)m * ldc + n] = v;
            }
        }
    }
}

// ---------------------------------------------------------------------------
// Deformable sampling: one wave per (b, q, head); lane = channel (DH=32).
// Branchless gather: clamp index, unconditional load, mask with select.
// ---------------------------------------------------------------------------
__device__ __forceinline__ float fetch_val(const float* __restrict__ value,
                                           int b, int st, int Wl, int Hl,
                                           int x, int y, int h, int lane)
{
    const bool valid = (x >= 0) & (x < Wl) & (y >= 0) & (y < Hl);
    const int cx = min(max(x, 0), Wl - 1);
    const int cy = min(max(y, 0), Hl - 1);
    const size_t idx = (size_t)b * kLIN + st + cy * Wl + cx;
    const float v = value[(idx * kNH + h) * kDH + lane];
    return valid ? v : 0.0f;
}

__global__ __launch_bounds__(256)
void msda_sample(const float* __restrict__ off, const float* __restrict__ logits,
                 const float* __restrict__ value, float* __restrict__ msda)
{
    const int gw   = (blockIdx.x * 256 + threadIdx.x) >> 5;   // (b*LQ+q)*NH+h
    const int lane = threadIdx.x & 31;
    if (gw >= kB * kLQ * kNH) return;
    const int h  = gw & 7;
    const int bq = gw >> 3;
    const int q  = bq & (kLQ - 1);
    const int b  = bq >> 10;

    const float refx = ((q & 31) + 0.5f) * (1.0f / 32.0f);
    const float refy = ((q >> 5) + 0.5f) * (1.0f / 32.0f);

    const float* ob = off + (size_t)gw * (kL * kP * 2);
    const float* lb = logits + (size_t)gw * (kL * kP);

    float aw[12];
    float mx = -1e30f;
#pragma unroll
    for (int i = 0; i < 12; ++i) { aw[i] = lb[i]; mx = fmaxf(mx, aw[i]); }
    float sm = 0.0f;
#pragma unroll
    for (int i = 0; i < 12; ++i) { aw[i] = __expf(aw[i] - mx); sm += aw[i]; }
    const float inv = 1.0f / sm;
#pragma unroll
    for (int i = 0; i < 12; ++i) aw[i] *= inv;

    const int starts[3] = {0, 4096, 5120};
    const int dims[3]   = {64, 32, 16};

    float acc = 0.0f;
#pragma unroll
    for (int lvl = 0; lvl < 3; ++lvl) {
        const int Wl = dims[lvl], Hl = dims[lvl], st = starts[lvl];
#pragma unroll
        for (int p = 0; p < 4; ++p) {
            const float ox = ob[(lvl * 4 + p) * 2 + 0];
            const float oy = ob[(lvl * 4 + p) * 2 + 1];
            const float lx = refx * Wl + ox - 0.5f;
            const float ly = refy * Hl + oy - 0.5f;
            const float fx = floorf(lx), fy = floorf(ly);
            const int x0 = (int)fx, y0 = (int)fy;
            const float dx = lx - fx, dy = ly - fy;
            const float a = aw[lvl * 4 + p];
            const float w00 = (1.0f - dx) * (1.0f - dy) * a;
            const float w10 = dx * (1.0f - dy) * a;
            const float w01 = (1.0f - dx) * dy * a;
            const float w11 = dx * dy * a;
            acc += w00 * fetch_val(value, b, st, Wl, Hl, x0,     y0,     h, lane);
            acc += w10 * fetch_val(value, b, st, Wl, Hl, x0 + 1, y0,     h, lane);
            acc += w01 * fetch_val(value, b, st, Wl, Hl, x0,     y0 + 1, h, lane);
            acc += w11 * fetch_val(value, b, st, Wl, Hl, x0 + 1, y0 + 1, h, lane);
        }
    }
    // (B, LQ, NH, DH) == (B, LQ, D) with d = h*32+lane
    msda[(size_t)gw * kDH + lane] = acc;
}

// ---------------------------------------------------------------------------
// Concept k/v projections: kv[m][c] = concepts[m,:]·Wkv[:,c] + bkv[c]
// sets: 0=unsup (Wkv_c), 1=concepts (Wkv_c), 2=spatial (Wkv_s)
// ---------------------------------------------------------------------------
__global__ __launch_bounds__(256)
void kv_proj(const float* __restrict__ cu, const float* __restrict__ cn,
             const float* __restrict__ cs,
             const float* __restrict__ Wkv_c, const float* __restrict__ bkv_c,
             const float* __restrict__ Wkv_s, const float* __restrict__ bkv_s,
             float* __restrict__ kvu, float* __restrict__ kvn, float* __restrict__ kvs)
{
    const int t = blockIdx.x * 256 + threadIdx.x;
    if (t >= 3 * kNC * 2 * kD) return;
    const int set = t / (kNC * 2 * kD);
    const int r   = t % (kNC * 2 * kD);
    const int m   = r / (2 * kD);
    const int c   = r % (2 * kD);
    const float* con = (set == 0) ? cu : (set == 1) ? cn : cs;
    const float* W   = (set == 2) ? Wkv_s : Wkv_c;
    const float* bb  = (set == 2) ? bkv_s : bkv_c;
    float s = bb[c];
    for (int k = 0; k < kD; ++k) s += con[m * kD + k] * W[k * 2 * kD + c];
    float* outp = (set == 0) ? kvu : (set == 1) ? kvn : kvs;
    outp[r] = s;
}

// ---------------------------------------------------------------------------
// Class-token concept attention (both unsup + named branches). 1 block per b.
// ---------------------------------------------------------------------------
__global__ __launch_bounds__(256)
void cls_attn(const float* __restrict__ x_cls, const float* __restrict__ Wq,
              const float* __restrict__ bq,
              const float* __restrict__ kvu, const float* __restrict__ kvn,
              const float* __restrict__ Wp, const float* __restrict__ bp,
              float* __restrict__ out_cls,
              float* __restrict__ attn_u_out, float* __restrict__ attn_n_out)
{
    __shared__ float q[256], o[256], lg[20], attn[20];
    const int b = blockIdx.x, t = threadIdx.x;

    float s = bq[t];
    for (int k = 0; k < kD; ++k) s += x_cls[b * kD + k] * Wq[k * kD + t];
    q[t] = s;
    __syncthreads();

    float outacc = 0.0f;
    for (int br = 0; br < 2; ++br) {
        const float* kv = br ? kvn : kvu;
        if (t < 20) {
            const int h = t / 10, m = t % 10;
            float d = 0.0f;
            for (int k = 0; k < 128; ++k) d += q[h * 128 + k] * kv[m * 512 + h * 128 + k];
            lg[t] = d * kInvSqrt128;
        }
        __syncthreads();
        if (t < 2) {
            float mx = -1e30f;
            for (int m = 0; m < 10; ++m) mx = fmaxf(mx, lg[t * 10 + m]);
            float sme = 0.0f;
            for (int m = 0; m < 10; ++m) { float e = __expf(lg[t * 10 + m] - mx); attn[t * 10 + m] = e; sme += e; }
            for (int m = 0; m < 10; ++m) attn[t * 10 + m] /= sme;
        }
        __syncthreads();
        {
            const int h = t >> 7, dd = t & 127;
            float v = 0.0f;
            for (int m = 0; m < 10; ++m) v += attn[h * 10 + m] * kv[m * 512 + 256 + h * 128 + dd];
            o[t] = v;
        }
        __syncthreads();
        if (t < kNCLS) {
            float v = bp[t];
            for (int d = 0; d < kD; ++d) v += o[d] * Wp[d * kNCLS + t];
            outacc += v;
        }
        if (t < 10) {
            const float a = 0.5f * (attn[t] + attn[10 + t]);
            (br ? attn_n_out : attn_u_out)[b * 10 + t] = a;
        }
        __syncthreads();
    }
    if (t < kNCLS) out_cls[b * kNCLS + t] = outacc;
}

// ---------------------------------------------------------------------------
// Spatial concept attention: one wave per (b, lq); loops over 2 heads.
// ---------------------------------------------------------------------------
__global__ __launch_bounds__(256)
void spatial_attn(const float* __restrict__ qs, const float* __restrict__ kvs,
                  float* __restrict__ os, float* __restrict__ attn_s_out)
{
    const int gw   = (blockIdx.x * 256 + threadIdx.x) >> 5;   // b*LQ + q
    const int lane = threadIdx.x & 31;
    if (gw >= kMQ) return;
    const float* qp = qs + (size_t)gw * kD;

    float am[10];
#pragma unroll
    for (int m = 0; m < 10; ++m) am[m] = 0.0f;

    for (int h = 0; h < 2; ++h) {
        float qv[4];
#pragma unroll
        for (int j = 0; j < 4; ++j) qv[j] = qp[h * 128 + lane + 32 * j];

        float a[10];
        float mx = -1e30f;
#pragma unroll
        for (int m = 0; m < 10; ++m) {
            float d = 0.0f;
#pragma unroll
            for (int j = 0; j < 4; ++j) d += qv[j] * kvs[m * 512 + h * 128 + lane + 32 * j];
            for (int o2 = 16; o2; o2 >>= 1) d += __shfl_xor(d, o2, 32);
            a[m] = d * kInvSqrt128;
            mx = fmaxf(mx, a[m]);
        }
        float sme = 0.0f;
#pragma unroll
        for (int m = 0; m < 10; ++m) { a[m] = __expf(a[m] - mx); sme += a[m]; }
        const float inv = 1.0f / sme;
#pragma unroll
        for (int m = 0; m < 10; ++m) { a[m] *= inv; am[m] += 0.5f * a[m]; }

#pragma unroll
        for (int j = 0; j < 4; ++j) {
            float ov = 0.0f;
#pragma unroll
            for (int m = 0; m < 10; ++m) ov += a[m] * kvs[m * 512 + 256 + h * 128 + lane + 32 * j];
            os[(size_t)gw * kD + h * 128 + lane + 32 * j] = ov;
        }
    }
#pragma unroll
    for (int m = 0; m < 10; ++m)
        if (lane == m) attn_s_out[(size_t)gw * 10 + m] = am[m];
}

// ---------------------------------------------------------------------------
// Final: out[b,c] = cls_branches[b,c] + mean_lq(outs[b,lq,c])
// ---------------------------------------------------------------------------
__global__ __launch_bounds__(256)
void finalize(const float* __restrict__ out_cls, const float* __restrict__ outs,
              float* __restrict__ out)
{
    const int b = blockIdx.x, t = threadIdx.x;
    if (t < kNCLS) {
        float s = 0.0f;
        for (int l = 0; l < kLQ; ++l) s += outs[((size_t)b * kLQ + l) * kNCLS + t];
        out[b * kNCLS + t] = out_cls[b * kNCLS + t] + s * (1.0f / (float)kLQ);
    }
}

// ---------------------------------------------------------------------------
extern "C" void kernel_launch(void* const* d_in, const int* in_sizes, int n_in,
                              void* d_out, int out_size, void* d_ws, size_t ws_size,
                              hipStream_t stream) {
    const float* query  = (const float*)d_in[0];
    const float* feats  = (const float*)d_in[1];
    const float* x_cls  = (const float*)d_in[2];
    const float* W_off  = (const float*)d_in[3];
    const float* b_off  = (const float*)d_in[4];
    const float* W_attn = (const float*)d_in[5];
    const float* b_attn = (const float*)d_in[6];
    const float* W_val  = (const float*)d_in[7];
    const float* b_val  = (const float*)d_in[8];
    const float* W_out  = (const float*)d_in[9];
    const float* b_out  = (const float*)d_in[10];
    const float* c_uns  = (const float*)d_in[11];
    const float* c_nam  = (const float*)d_in[12];
    const float* c_spa  = (const float*)d_in[13];
    const float* Wq_c   = (const float*)d_in[14];
    const float* bq_c   = (const float*)d_in[15];
    const float* Wkv_c  = (const float*)d_in[16];
    const float* bkv_c  = (const float*)d_in[17];
    const float* Wp_c   = (const float*)d_in[18];
    const float* bp_c   = (const float*)d_in[19];
    const float* Wq_s   = (const float*)d_in[20];
    const float* bq_s   = (const float*)d_in[21];
    const float* Wkv_s  = (const float*)d_in[22];
    const float* bkv_s  = (const float*)d_in[23];
    const float* Wp_s   = (const float*)d_in[24];
    const float* bp_s   = (const float*)d_in[25];

    float* w = (float*)d_ws;
    float* ws_value = w;                                   // 172032*256
    float* ws_off   = ws_value + (size_t)kMV * kD;         // 32768*192
    float* ws_log   = ws_off   + (size_t)kMQ * 192;        // 32768*96
    float* ws_msda  = ws_log   + (size_t)kMQ * 96;         // 32768*256
    float* ws_x     = ws_msda  + (size_t)kMQ * kD;         // 32768*256
    float* ws_qs    = ws_x     + (size_t)kMQ * kD;         // 32768*256
    float* ws_os    = ws_qs    + (size_t)kMQ * kD;         // 32768*256
    float* ws_outs  = ws_os    + (size_t)kMQ * kD;         // 32768*200
    float* ws_kvu   = ws_outs  + (size_t)kMQ * kNCLS;      // 10*512
    float* ws_kvn   = ws_kvu   + kNC * 2 * kD;
    float* ws_kvs   = ws_kvn   + kNC * 2 * kD;
    float* ws_ocls  = ws_kvs   + kNC * 2 * kD;             // 32*200

    float* out      = (float*)d_out;                       // (32,200)
    float* attn_u   = out + kB * kNCLS;                    // (32,10)
    float* attn_n   = attn_u + kB * kNC;                   // (32,10)
    float* attn_s   = attn_n + kB * kNC;                   // (32,1024,10)

    const dim3 blk128(128), blk256(256);

    // 1) value = feats @ W_val + b_val          (N=256 -> NT=16, A read once)
    gemm_strip<16><<<dim3(kMV / 64), blk128, 0, stream>>>(
        feats, W_val, b_val, nullptr, ws_value, kMV, kD, kD, kD);
    // 2) off = query @ W_off + b_off            (N=192 -> NT=12)
    gemm_strip<12><<<dim3(kMQ / 64), blk128, 0, stream>>>(
        query, W_off, b_off, nullptr, ws_off, kMQ, 192, kD, 192);
    // 3) attn logits = query @ W_attn + b_attn  (N=96 -> NT=6)
    gemm_strip<6><<<dim3(kMQ / 64), blk128, 0, stream>>>(
        query, W_attn, b_attn, nullptr, ws_log, kMQ, 96, kD, 96);
    // 4) concept k/v projections (independent)
    kv_proj<<<dim3((3 * kNC * 2 * kD + 255) / 256), blk256, 0, stream>>>(
        c_uns, c_nam, c_spa, Wkv_c, bkv_c, Wkv_s, bkv_s, ws_kvu, ws_kvn, ws_kvs);
    // 5) deformable sampling -> msda (B,LQ,D)
    msda_sample<<<dim3((kB * kLQ * kNH * 32) / 256), blk256, 0, stream>>>(
        ws_off, ws_log, ws_value, ws_msda);
    // 6) x = query + msda @ W_out + b_out       (N=256 -> NT=16)
    gemm_strip<16><<<dim3(kMQ / 64), blk128, 0, stream>>>(
        ws_msda, W_out, b_out, query, ws_x, kMQ, kD, kD, kD);
    // 7) q_s = x @ Wq_s + bq_s                  (N=256 -> NT=16)
    gemm_strip<16><<<dim3(kMQ / 64), blk128, 0, stream>>>(
        ws_x, Wq_s, bq_s, nullptr, ws_qs, kMQ, kD, kD, kD);
    // 8) spatial concept attention -> o_s, attn_s (head-mean)
    spatial_attn<<<dim3((kMQ * 32) / 256), blk256, 0, stream>>>(
        ws_qs, ws_kvs, ws_os, attn_s);
    // 9) outs = o_s @ Wp_s + bp_s               (N=200 -> NT=13, padded cols)
    gemm_strip<13><<<dim3(kMQ / 64), blk128, 0, stream>>>(
        ws_os, Wp_s, bp_s, nullptr, ws_outs, kMQ, kNCLS, kD, kNCLS);
    // 10) class-token concept branches (unsup + named)
    cls_attn<<<dim3(kB), blk256, 0, stream>>>(
        x_cls, Wq_c, bq_c, ws_kvu, ws_kvn, Wp_c, bp_c, ws_ocls, attn_u, attn_n);
    // 11) out = cls branches + mean_lq(outs)
    finalize<<<dim3(kB), blk256, 0, stream>>>(ws_ocls, ws_outs, out);
}
